// LSTMModelInt8QAT_41214506172755
// MI455X (gfx1250) — 8-bit matrix operand order repaired and checked by reading, compiled, not run on hardware
//
#include <hip/hip_runtime.h>
#include <hip/hip_bf16.h>


typedef __attribute__((ext_vector_type(8))) int v8i;

#define SQ      0.05f
#define INV_SQ  20.0f
#define SQ2     0.0025f
#define HID     512
#define SEQ     512
#define DIN     64
#define NBATCH  256

#define OFF_IH0   0u
#define OFF_HH0   131072u
#define OFF_IH1   1179648u
#define OFF_HH1   2228224u
#define OFF_XQ    3276800u
#define OFF_Y     11665408u

__device__ __forceinline__ float fqv(float x) {
  float q = fminf(fmaxf(rintf(x * INV_SQ), -127.f), 127.f);
  return q * SQ;
}
__device__ __forceinline__ float ste_sig(float x) {
  float y = 1.f / (1.f + __expf(-x));
  return rintf(y * 256.f) * (1.f / 256.f);
}
__device__ __forceinline__ float ste_tanh(float x) {
  float y = tanhf(x);
  return rintf(y * 256.f) * (1.f / 256.f);
}

__launch_bounds__(256)
__global__ void pack_w_kernel(const float* __restrict__ W, signed char* __restrict__ dst,
                              int Kdim, int KT, int total) {
  int p = blockIdx.x * blockDim.x + threadIdx.x;
  if (p >= total) return;
  int blk  = p >> 10;
  int inb  = p & 1023;
  int lane = inb >> 5;
  int boff = inb & 31;
  int v = boff >> 2, j = boff & 3;
  int nt = blk / KT, kt = blk % KT;
  int l16 = lane & 15, hi = lane >> 4;
  int K = (v >> 1) * 16 + hi * 8 + (v & 1) * 4 + j;
  int n = nt * 16 + l16;
  int k = kt * 64 + K;
  float q = fminf(fmaxf(rintf(W[(size_t)n * Kdim + k] * INV_SQ), -127.f), 127.f);
  dst[p] = (signed char)(int)q;
}

__launch_bounds__(256)
__global__ void quant_x_kernel(const float* __restrict__ x, signed char* __restrict__ xq,
                               int total) {
  int p = blockIdx.x * blockDim.x + threadIdx.x;
  if (p >= total) return;
  float q = fminf(fmaxf(rintf(x[p] * INV_SQ), -127.f), 127.f);
  xq[p] = (signed char)(int)q;
}

__launch_bounds__(1024, 1)
__global__ void qlstm_kernel(const signed char* __restrict__ xq,
                             const signed char* __restrict__ wpk,
                             const float* __restrict__ bih0, const float* __restrict__ bhh0,
                             const float* __restrict__ bih1, const float* __restrict__ bhh1,
                             float* __restrict__ Y) {
  __shared__ signed char h0q[2][16][HID];
  __shared__ signed char h1q[2][16][HID];

  const int tid  = threadIdx.x;
  const int wv   = tid >> 5;
  const int lane = tid & 31;
  const int l16  = lane & 15;
  const int hi   = lane >> 4;
  const int b0   = blockIdx.x * 16;
  const int n0   = wv * 16;

  for (int i = tid; i < 2 * 16 * HID / 4; i += blockDim.x) {
    ((int*)h0q)[i] = 0;
    ((int*)h1q)[i] = 0;
  }
  __syncthreads();

  float bs0[4], bs1[4];
#pragma unroll
  for (int g = 0; g < 4; ++g) {
    int gcol = g * HID + n0 + l16;
    bs0[g] = bih0[gcol] + bhh0[gcol];
    bs1[g] = bih1[gcol] + bhh1[gcol];
  }

  float c0r[8], c1r[8];
#pragma unroll
  for (int e = 0; e < 8; ++e) { c0r[e] = 0.f; c1r[e] = 0.f; }

  auto loadA = [&](const signed char* buf, int k0) -> v8i {
    v8i a;
#pragma unroll
    for (int c = 0; c < 4; ++c) {
      int2 q = *(const int2*)(buf + l16 * HID + k0 + hi * 8 + c * 16);
      a[2 * c] = q.x; a[2 * c + 1] = q.y;
    }
    return a;
  };
  auto loadB = [&](const signed char* base, unsigned byteoff) -> v8i {
    const int4* p = (const int4*)(base + byteoff + lane * 32);
    int4 q0 = p[0], q1 = p[1];
    v8i b;
    b[0] = q0.x; b[1] = q0.y; b[2] = q0.z; b[3] = q0.w;
    b[4] = q1.x; b[5] = q1.y; b[6] = q1.z; b[7] = q1.w;
    return b;
  };

  const v8i vzero = {0, 0, 0, 0, 0, 0, 0, 0};
  int ping = 0;

  unsigned int zoff = 0;

#pragma unroll 1
  for (int t = 0; t < SEQ; ++t) {
    const int pong = ping ^ 1;

    asm volatile("" : "+s"(zoff));
    const signed char* wb = wpk + zoff;

    if (t + 1 < SEQ)
      __builtin_prefetch(xq + ((size_t)(b0 + l16) * SEQ + (t + 1)) * DIN, 0, 1);

    v8i acc[4];
#pragma unroll
    for (int g = 0; g < 4; ++g) acc[g] = vzero;

    {
      v8i ax;
#pragma unroll
      for (int c = 0; c < 4; ++c) {
        int2 q = *(const int2*)(xq + ((size_t)(b0 + l16) * SEQ + t) * DIN + hi * 8 + c * 16);
        ax[2 * c] = q.x; ax[2 * c + 1] = q.y;
      }
#pragma unroll
      for (int g = 0; g < 4; ++g) {
        v8i bm = loadB(wb, OFF_IH0 + (unsigned)(g * 32 + wv) * 1024u);
        acc[g] = __builtin_amdgcn_wmma_i32_16x16x64_iu8(true, ax, true, bm, acc[g],
                                                        false, false);
      }
    }
#pragma unroll
    for (int kt = 0; kt < 8; ++kt) {
      v8i ah = loadA(&h0q[ping][0][0], kt * 64);
#pragma unroll
      for (int g = 0; g < 4; ++g) {
        v8i bm = loadB(wb, OFF_HH0 + (unsigned)((g * 32 + wv) * 8 + kt) * 1024u);
        acc[g] = __builtin_amdgcn_wmma_i32_16x16x64_iu8(true, ah, true, bm, acc[g],
                                                        false, false);
      }
    }
#pragma unroll
    for (int e = 0; e < 8; ++e) {
      float iv = ste_sig (SQ2 * (float)acc[0][e] + bs0[0]);
      float fv = ste_sig (SQ2 * (float)acc[1][e] + bs0[1]);
      float gv = ste_tanh(SQ2 * (float)acc[2][e] + bs0[2]);
      float ov = ste_sig (SQ2 * (float)acc[3][e] + bs0[3]);
      float c  = fminf(fmaxf(fv * c0r[e] + iv * gv, -8.f), 8.f);
      c0r[e]   = c;
      float h  = ov * ste_tanh(c);
      float q  = fminf(fmaxf(rintf(h * INV_SQ), -127.f), 127.f);
      h0q[pong][hi * 8 + e][n0 + l16] = (signed char)(int)q;
    }
    __syncthreads();

#pragma unroll
    for (int g = 0; g < 4; ++g) acc[g] = vzero;
#pragma unroll
    for (int kt = 0; kt < 8; ++kt) {
      v8i ah = loadA(&h0q[pong][0][0], kt * 64);
#pragma unroll
      for (int g = 0; g < 4; ++g) {
        v8i bm = loadB(wb, OFF_IH1 + (unsigned)((g * 32 + wv) * 8 + kt) * 1024u);
        acc[g] = __builtin_amdgcn_wmma_i32_16x16x64_iu8(true, ah, true, bm, acc[g],
                                                        false, false);
      }
    }
#pragma unroll
    for (int kt = 0; kt < 8; ++kt) {
      v8i ah = loadA(&h1q[ping][0][0], kt * 64);
#pragma unroll
      for (int g = 0; g < 4; ++g) {
        v8i bm = loadB(wb, OFF_HH1 + (unsigned)((g * 32 + wv) * 8 + kt) * 1024u);
        acc[g] = __builtin_amdgcn_wmma_i32_16x16x64_iu8(true, ah, true, bm, acc[g],
                                                        false, false);
      }
    }
#pragma unroll
    for (int e = 0; e < 8; ++e) {
      float iv = ste_sig (SQ2 * (float)acc[0][e] + bs1[0]);
      float fv = ste_sig (SQ2 * (float)acc[1][e] + bs1[1]);
      float gv = ste_tanh(SQ2 * (float)acc[2][e] + bs1[2]);
      float ov = ste_sig (SQ2 * (float)acc[3][e] + bs1[3]);
      float c  = fminf(fmaxf(fv * c1r[e] + iv * gv, -8.f), 8.f);
      c1r[e]   = c;
      float h  = ov * ste_tanh(c);
      int row  = hi * 8 + e;
      Y[((size_t)(b0 + row) * SEQ + t) * HID + n0 + l16] = h;
      float q  = fminf(fmaxf(rintf(h * INV_SQ), -127.f), 127.f);
      h1q[pong][row][n0 + l16] = (signed char)(int)q;
    }
    __syncthreads();
    ping = pong;
  }
}

__launch_bounds__(256)
__global__ void attn_fc_kernel(const float* __restrict__ Y,
                               const float* __restrict__ Wfc,
                               const float* __restrict__ bfc,
                               float* __restrict__ out) {
  const int b = blockIdx.x;
  const int tid = threadIdx.x;
  __shared__ float hl[HID];
  __shared__ float sc[SEQ];
  __shared__ float red[256];
  const float* Yb = Y + (size_t)b * SEQ * HID;

  for (int h = tid; h < HID; h += 256) hl[h] = Yb[(size_t)(SEQ - 1) * HID + h];
  __syncthreads();

  for (int w = tid; w < SEQ; w += 256) {
    const float* yr = Yb + (size_t)w * HID;
    float s = 0.f;
    for (int h = 0; h < HID; ++h) s += yr[h] * hl[h];
    sc[w] = s;
  }
  __syncthreads();

  red[tid] = fmaxf(sc[tid], sc[tid + 256]);
  __syncthreads();
  for (int off = 128; off > 0; off >>= 1) {
    if (tid < off) red[tid] = fmaxf(red[tid], red[tid + off]);
    __syncthreads();
  }
  float mx = red[0];
  __syncthreads();

  float e0 = __expf(sc[tid] - mx), e1 = __expf(sc[tid + 256] - mx);
  sc[tid] = e0; sc[tid + 256] = e1;
  red[tid] = e0 + e1;
  __syncthreads();
  for (int off = 128; off > 0; off >>= 1) {
    if (tid < off) red[tid] += red[tid + off];
    __syncthreads();
  }
  float inv = 1.f / red[0];
  __syncthreads();

  float part = 0.f;
  for (int h = tid; h < HID; h += 256) {
    float acc = 0.f;
    for (int w = 0; w < SEQ; ++w) acc += sc[w] * Yb[(size_t)w * HID + h];
    float ctx = acc * inv;
    part += fqv(hl[h]) * fqv(Wfc[h]) + fqv(ctx) * fqv(Wfc[HID + h]);
  }
  red[tid] = part;
  __syncthreads();
  for (int off = 128; off > 0; off >>= 1) {
    if (tid < off) red[tid] += red[tid + off];
    __syncthreads();
  }
  if (tid == 0) out[b] = red[0] + bfc[0];
}

extern "C" void kernel_launch(void* const* d_in, const int* in_sizes, int n_in,
                              void* d_out, int out_size, void* d_ws, size_t ws_size,
                              hipStream_t stream) {
  const float* x    = (const float*)d_in[0];
  const float* Wih0 = (const float*)d_in[1];
  const float* bih0 = (const float*)d_in[2];
  const float* Whh0 = (const float*)d_in[3];
  const float* bhh0 = (const float*)d_in[4];
  const float* Wih1 = (const float*)d_in[5];
  const float* bih1 = (const float*)d_in[6];
  const float* Whh1 = (const float*)d_in[7];
  const float* bhh1 = (const float*)d_in[8];
  const float* Wfc  = (const float*)d_in[9];
  const float* bfc  = (const float*)d_in[10];
  float* out = (float*)d_out;

  char* ws = (char*)d_ws;
  signed char* wpk  = (signed char*)(ws + OFF_IH0);
  signed char* xq   = (signed char*)(ws + OFF_XQ);
  float*       Y    = (float*)(ws + OFF_Y);

  pack_w_kernel<<<(131072 + 255) / 256, 256, 0, stream>>>(Wih0, wpk + OFF_IH0, DIN, 1, 131072);
  pack_w_kernel<<<(1048576 + 255) / 256, 256, 0, stream>>>(Whh0, wpk + OFF_HH0, HID, 8, 1048576);
  pack_w_kernel<<<(1048576 + 255) / 256, 256, 0, stream>>>(Wih1, wpk + OFF_IH1, HID, 8, 1048576);
  pack_w_kernel<<<(1048576 + 255) / 256, 256, 0, stream>>>(Whh1, wpk + OFF_HH1, HID, 8, 1048576);

  quant_x_kernel<<<(NBATCH * SEQ * DIN + 255) / 256, 256, 0, stream>>>(
      x, xq, NBATCH * SEQ * DIN);

  qlstm_kernel<<<dim3(NBATCH / 16), dim3(1024), 0, stream>>>(
      xq, wpk, bih0, bhh0, bih1, bhh1, Y);

  attn_fc_kernel<<<dim3(NBATCH), dim3(256), 0, stream>>>(Y, Wfc, bfc, out);
}
